// Mask_Attention_66125316489673
// MI455X (gfx1250) — matrix operand order repaired and checked by reading, not compiled, not run on hardware
//
#include <hip/hip_runtime.h>
#include <hip/hip_bf16.h>
#include <math.h>

// ---------------------------------------------------------------------------
// Problem constants
// ---------------------------------------------------------------------------
#define BT      40          // b * T
#define NTOK    3137        // tokens per bt (1 A-token + 3136 V-tokens)
#define NV      3136
#define DIM     512
#define TT      5           // T
#define BB      8           // batch b
#define HH      8           // heads
#define DH      64          // dim head
#define INNER   512
#define KV      (TT * NV)   // 15680 keys for attention-1
#define MROWS_V (BT * NV)   // 125440
#define MROWS_O (BT * NTOK) // 125480
#define SCALE   0.125f      // 64^-0.5

typedef __attribute__((ext_vector_type(16))) __bf16 v16bf;
typedef __attribute__((ext_vector_type(8)))  float  v8f;
typedef __attribute__((ext_vector_type(4)))  int    v4i;

// ---------------------------------------------------------------------------
// Async global->LDS copy (gfx1250 ASYNCcnt path), with safe fallback
// Builtin signature (from compiler diagnostic): param0 = v4i addrspace(1)*,
// i.e. (global <4 x i32>*, lds <4 x i32>*, imm offset, imm cpol)
// ---------------------------------------------------------------------------
#if __has_builtin(__builtin_amdgcn_global_load_async_to_lds_b128)
#define HAS_ASYNC_LDS 1
#else
#define HAS_ASYNC_LDS 0
#endif

typedef __attribute__((address_space(1))) v4i GV4;
typedef __attribute__((address_space(3))) v4i LV4;

__device__ __forceinline__ void async_copy16(const __bf16* g, __bf16* l) {
#if HAS_ASYNC_LDS
    __builtin_amdgcn_global_load_async_to_lds_b128(
        (GV4*)(void*)g, (LV4*)(void*)l, 0, 0);
#else
    *(uint4*)l = *(const uint4*)g;
#endif
}

__device__ __forceinline__ void wait_async() {
#if HAS_ASYNC_LDS
#if __has_builtin(__builtin_amdgcn_s_wait_asynccnt)
    __builtin_amdgcn_s_wait_asynccnt(0);
#else
    asm volatile("s_wait_asynccnt 0" ::: "memory");
#endif
#endif
}

// ---------------------------------------------------------------------------
// WMMA fragment loaders from LDS tiles (bf16, 16x16x32)
// A 16x32 (MxK): lane m=lane%16, holds K = {8h..8h+7} U {16+8h..16+8h+7}
// B 32x16 (KxN) from row-major Bt (N x K): lane n=lane%16, holds K = {8h..8h+7} U {16+8h..16+8h+7}
// ---------------------------------------------------------------------------
__device__ __forceinline__ v16bf lds_frag_a(const __bf16* row, int half) {
    union { uint4 u[2]; v16bf v; } f;
    f.u[0] = *(const uint4*)(row + 8 * half);
    f.u[1] = *(const uint4*)(row + 16 + 8 * half);
    return f.v;
}
__device__ __forceinline__ v16bf lds_frag_b(const __bf16* row, int half) {
    union { uint4 u[2]; v16bf v; } f;
    f.u[0] = *(const uint4*)(row + 8 * half);
    f.u[1] = *(const uint4*)(row + 16 + 8 * half);
    return f.v;
}

// ---------------------------------------------------------------------------
// Block-cooperative bf16 WMMA GEMM: C[M,N] = A[M,K] @ Bt[N,K]^T (+bias)
// 128x128 block tile, 4 waves (each 64x64), K-step 32, double-buffered LDS
// staged via async global->LDS loads.
// a_mode==1 : A rows are V rows of xb -> physical row = r + r/3136 + 1
// c_f32==1  : write float output, else bf16
// ---------------------------------------------------------------------------
__global__ __launch_bounds__(128) void gemm_bf16_wmma(
    const __bf16* __restrict__ A, const __bf16* __restrict__ Bt,
    void* __restrict__ Cout, const float* __restrict__ bias,
    int M, int N, int K, int ldc, int a_mode, int c_f32)
{
    __shared__ __attribute__((aligned(16))) __bf16 smA[2][128 * 32];
    __shared__ __attribute__((aligned(16))) __bf16 smB[2][128 * 32];

    const int tid  = threadIdx.x;
    const int lane = tid & 31;
    const int wave = tid >> 5;
    const int wy = wave >> 1, wx = wave & 1;      // 2x2 wave grid
    const int mloc = lane & 15;
    const int half = lane >> 4;
    const int row0 = blockIdx.x * 128;
    const int col0 = blockIdx.y * 128;

    // Staging addresses: thread covers 4 chunks of 16B for A and 4 for B.
    // chunk c = q*128 + tid : row = 32*q + (tid>>2), seg = tid&3 (8 elems)
    size_t g_aoff[4], g_boff[4];
#pragma unroll
    for (int q = 0; q < 4; ++q) {
        int r = row0 + 32 * q + (tid >> 2);
        if (r >= M) r = M - 1;                    // clamp (valid mem; store guarded)
        size_t ar = (size_t)r;
        if (a_mode == 1) ar = (size_t)r + (size_t)(r / NV) + 1;
        g_aoff[q] = ar * (size_t)K + (size_t)((tid & 3) * 8);
        int n = col0 + 32 * q + (tid >> 2);       // N multiple of 128
        g_boff[q] = (size_t)n * (size_t)K + (size_t)((tid & 3) * 8);
    }
    const int l_off0 = (tid >> 2) * 32 + (tid & 3) * 8;

    v8f acc[4][4] = {};

    // preload stage 0
#pragma unroll
    for (int q = 0; q < 4; ++q) {
        async_copy16(A + g_aoff[q], &smA[0][l_off0 + q * 32 * 32]);
        async_copy16(Bt + g_boff[q], &smB[0][l_off0 + q * 32 * 32]);
    }
    wait_async();
    __syncthreads();

    int buf = 0;
    for (int k0 = 0; k0 < K; k0 += 32) {
        const int nbuf = buf ^ 1;
        if (k0 + 32 < K) {
#pragma unroll
            for (int q = 0; q < 4; ++q) {
                async_copy16(A + g_aoff[q] + k0 + 32, &smA[nbuf][l_off0 + q * 32 * 32]);
                async_copy16(Bt + g_boff[q] + k0 + 32, &smB[nbuf][l_off0 + q * 32 * 32]);
            }
        }

        const __bf16* As = smA[buf];
        const __bf16* Bs = smB[buf];
        v16bf af[4], bfm[4];
#pragma unroll
        for (int i = 0; i < 4; ++i)
            af[i] = lds_frag_a(As + (64 * wy + 16 * i + mloc) * 32, half);
#pragma unroll
        for (int j = 0; j < 4; ++j)
            bfm[j] = lds_frag_b(Bs + (64 * wx + 16 * j + mloc) * 32, half);
#pragma unroll
        for (int i = 0; i < 4; ++i)
#pragma unroll
            for (int j = 0; j < 4; ++j)
                acc[i][j] = __builtin_amdgcn_wmma_f32_16x16x32_bf16(
                    false, af[i], false, bfm[j], (short)0, acc[i][j], false, false);

        wait_async();
        __syncthreads();
        buf = nbuf;
    }

    // store: lane holds column n = mloc, rows v + 8*half
#pragma unroll
    for (int i = 0; i < 4; ++i) {
#pragma unroll
        for (int v = 0; v < 8; ++v) {
            int m = row0 + 64 * wy + 16 * i + v + 8 * half;
            if (m >= M) continue;
            size_t crow = (size_t)m * (size_t)ldc;
#pragma unroll
            for (int j = 0; j < 4; ++j) {
                int n = col0 + 64 * wx + 16 * j + mloc;
                float val = acc[i][j][v];
                if (bias) val += bias[n];
                if (c_f32) ((float*)Cout)[crow + n] = val;
                else       ((__bf16*)Cout)[crow + n] = (__bf16)val;
            }
        }
    }
}

// ---------------------------------------------------------------------------
// Elementwise f32 -> bf16
// ---------------------------------------------------------------------------
__global__ void cvt_f32_bf16(const float* __restrict__ in, __bf16* __restrict__ out,
                             size_t n)
{
    size_t i = (size_t)blockIdx.x * blockDim.x + threadIdx.x;
    if (i < n) out[i] = (__bf16)in[i];
}

// Transpose+convert weight W[K,N] (f32) -> Wt[N,K] (bf16)
__global__ void wt_transpose(const float* __restrict__ W, __bf16* __restrict__ Wt,
                             int K, int N)
{
    size_t idx = (size_t)blockIdx.x * blockDim.x + threadIdx.x;
    if (idx < (size_t)K * N) {
        int n = (int)(idx / K), k = (int)(idx % K);
        Wt[idx] = (__bf16)W[(size_t)k * N + n];
    }
}

// ---------------------------------------------------------------------------
// aqkv = A_tokens @ w_aqkv  (40 x 512 @ 512 x 1536), f32 for accuracy
// grid (BT, 6), block 256
// ---------------------------------------------------------------------------
__global__ void aqkv_gemm(const float* __restrict__ x, const float* __restrict__ w,
                          float* __restrict__ aqkv)
{
    __shared__ float xr[DIM];
    int bt = blockIdx.x, tid = threadIdx.x;
    const float* xrow = x + (size_t)bt * NTOK * DIM;
    xr[tid]       = xrow[tid];
    xr[tid + 256] = xrow[tid + 256];
    __syncthreads();
    int j = blockIdx.y * 256 + tid;
    float acc = 0.f;
    for (int d = 0; d < DIM; ++d) acc += xr[d] * w[(size_t)d * 1536 + j];
    aqkv[(size_t)bt * 1536 + j] = acc;
}

// ---------------------------------------------------------------------------
// Attention 1 scores: dots[(b,h,q), k] = scale * qa . Kv_k
// grid (B*H, ceil(KV/256)), block 256
// ---------------------------------------------------------------------------
__global__ void att1_dots(const float* __restrict__ aqkv, const __bf16* __restrict__ vkv,
                          float* __restrict__ dots)
{
    __shared__ float qs[TT * DH];  // 5 x 64
    int bh = blockIdx.x, b = bh / HH, h = bh % HH, tid = threadIdx.x;
    for (int i = tid; i < TT * DH; i += 256) {
        int q = i / DH, d = i % DH;
        qs[i] = aqkv[(size_t)(b * TT + q) * 1536 + h * DH + d];
    }
    __syncthreads();
    int k = blockIdx.y * 256 + tid;
    if (k >= KV) return;
    const __bf16* kp = vkv + (size_t)(b * KV + k) * 1024 + h * DH; // vk half
    float acc[TT] = {};
    for (int d = 0; d < DH; ++d) {
        float kvd = (float)kp[d];
#pragma unroll
        for (int q = 0; q < TT; ++q) acc[q] += kvd * qs[q * DH + d];
    }
#pragma unroll
    for (int q = 0; q < TT; ++q)
        dots[((size_t)bh * TT + q) * KV + k] = acc[q] * SCALE;
}

// Row softmax over KV, in place. grid = 320 rows, block 256
__global__ void att1_softmax(float* __restrict__ dots)
{
    __shared__ float red[256];
    float* row = dots + (size_t)blockIdx.x * KV;
    int tid = threadIdx.x;
    float m = -INFINITY;
    for (int k = tid; k < KV; k += 256) m = fmaxf(m, row[k]);
    red[tid] = m; __syncthreads();
    for (int s = 128; s > 0; s >>= 1) { if (tid < s) red[tid] = fmaxf(red[tid], red[tid + s]); __syncthreads(); }
    m = red[0]; __syncthreads();
    float sum = 0.f;
    for (int k = tid; k < KV; k += 256) sum += expf(row[k] - m);
    red[tid] = sum; __syncthreads();
    for (int s = 128; s > 0; s >>= 1) { if (tid < s) red[tid] += red[tid + s]; __syncthreads(); }
    float inv = 1.0f / red[0];
    for (int k = tid; k < KV; k += 256) row[k] = expf(row[k] - m) * inv;
}

// av_o: out[q,d] = sum_k p[q,k] * Vv[k,d]; writes a_out rows into proj_in.
// grid = B*H, block 320 (q = tid/64, d = tid%64)
__global__ __launch_bounds__(320) void att1_out(const float* __restrict__ probs,
                                                const __bf16* __restrict__ vkv,
                                                __bf16* __restrict__ proj)
{
    int bh = blockIdx.x, b = bh / HH, h = bh % HH;
    int q = threadIdx.x / DH, d = threadIdx.x % DH;
    const float* prow = probs + ((size_t)bh * TT + q) * KV;
    const __bf16* vbase = vkv + (size_t)b * KV * 1024 + 512 + h * DH + d; // vv half
    float acc = 0.f;
    for (int k = 0; k < KV; ++k)
        acc += prow[k] * (float)vbase[(size_t)k * 1024];
    int bt = b * TT + q;
    proj[(size_t)bt * NTOK * DIM + h * DH + d] = (__bf16)acc; // token 0 row
}

// ---------------------------------------------------------------------------
// Attention 2: each V row attends to 5 A-keys per head.
// Pass 1: softmax weights. grid 3920, block 256 (pair = (r,h))
// ---------------------------------------------------------------------------
__global__ void att2_weights(const __bf16* __restrict__ vq, const float* __restrict__ aqkv,
                             float* __restrict__ vaw)
{
    __shared__ float aks[TT * INNER];  // ak for this b: 5 x 512
    size_t pair = (size_t)blockIdx.x * 256 + threadIdx.x;
    int r = (int)(pair >> 3), h = (int)(pair & 7);
    int b = r / KV;  // whole block shares b (125440*8 % 256 == 0 per b)
    for (int i = threadIdx.x; i < TT * INNER; i += 256) {
        int t = i / INNER, c = i % INNER;
        aks[i] = aqkv[(size_t)(b * TT + t) * 1536 + 512 + c];
    }
    __syncthreads();

    const __bf16* qp = vq + (size_t)r * INNER + h * DH;
    float qf[DH];
#pragma unroll
    for (int d = 0; d < DH; ++d) qf[d] = (float)qp[d];

    float dt[TT];
#pragma unroll
    for (int t = 0; t < TT; ++t) {
        float acc = 0.f;
        for (int d = 0; d < DH; ++d) acc += qf[d] * aks[t * INNER + h * DH + d];
        dt[t] = acc * SCALE;
    }
    float m = dt[0];
#pragma unroll
    for (int t = 1; t < TT; ++t) m = fmaxf(m, dt[t]);
    float s = 0.f;
#pragma unroll
    for (int t = 0; t < TT; ++t) { dt[t] = expf(dt[t] - m); s += dt[t]; }
    float inv = 1.0f / s;
#pragma unroll
    for (int t = 0; t < TT; ++t)
        vaw[(size_t)r * 40 + h * TT + t] = dt[t] * inv;
}

// Pass 2: weighted sum of Va, masked, write into proj_in. grid MROWS_V, block 512
__global__ void att2_out(const float* __restrict__ vaw, const float* __restrict__ aqkv,
                         const float* __restrict__ mask, __bf16* __restrict__ proj)
{
    __shared__ float avs[TT * INNER];
    __shared__ float ws[40];
    int r = blockIdx.x, tid = threadIdx.x;
    int b = r / KV;
    for (int i = tid; i < TT * INNER; i += 512) {
        int t = i / INNER, c = i % INNER;
        avs[i] = aqkv[(size_t)(b * TT + t) * 1536 + 1024 + c];
    }
    if (tid < 40) ws[tid] = vaw[(size_t)r * 40 + tid];
    __syncthreads();

    int c = tid, h = c >> 6;
    float acc = 0.f;
#pragma unroll
    for (int t = 0; t < TT; ++t) acc += ws[h * TT + t] * avs[t * INNER + c];
    float mv = mask[r];                       // mask flat (40,3136) == V rows
    int bt = r / NV;
    size_t rowOut = (size_t)r + bt + 1;       // bt*3137 + i + 1
    proj[rowOut * DIM + c] = (__bf16)(acc * mv);
}

// ---------------------------------------------------------------------------
// Host-side launcher
// ---------------------------------------------------------------------------
extern "C" void kernel_launch(void* const* d_in, const int* in_sizes, int n_in,
                              void* d_out, int out_size, void* d_ws, size_t ws_size,
                              hipStream_t stream)
{
    const float* x      = (const float*)d_in[0];
    const float* mask   = (const float*)d_in[1];
    const float* w_vq   = (const float*)d_in[2];
    const float* w_vkv  = (const float*)d_in[3];
    const float* w_aqkv = (const float*)d_in[4];
    const float* w_out  = (const float*)d_in[5];
    const float* b_out  = (const float*)d_in[6];
    float* out = (float*)d_out;

    char* ws = (char*)d_ws;
    size_t off = 0;
    auto alloc = [&](size_t bytes) { void* p = ws + off; off = (off + bytes + 255) & ~(size_t)255; return p; };

    __bf16* xb      = (__bf16*)alloc((size_t)BT * NTOK * DIM * 2);   // x as bf16
    __bf16* wt_vkv  = (__bf16*)alloc((size_t)1024 * DIM * 2);
    __bf16* wt_vq   = (__bf16*)alloc((size_t)512 * DIM * 2);
    __bf16* wt_out  = (__bf16*)alloc((size_t)512 * DIM * 2);
    __bf16* vkv     = (__bf16*)alloc((size_t)MROWS_V * 1024 * 2);    // [vk | vv]
    __bf16* vq      = (__bf16*)alloc((size_t)MROWS_V * 512 * 2);
    float*  aqkv    = (float*) alloc((size_t)BT * 1536 * 4);
    float*  dots    = (float*) alloc((size_t)BB * HH * TT * KV * 4);
    float*  vaw     = (float*) alloc((size_t)MROWS_V * 40 * 4);
    __bf16* proj    = (__bf16*)alloc((size_t)MROWS_O * DIM * 2);

    // 1) convert x -> bf16
    {
        size_t n = (size_t)BT * NTOK * DIM;
        cvt_f32_bf16<<<(unsigned)((n + 255) / 256), 256, 0, stream>>>(x, xb, n);
    }
    // 2) transpose + convert weights
    wt_transpose<<<(unsigned)(((size_t)512 * 1024 + 255) / 256), 256, 0, stream>>>(w_vkv, wt_vkv, 512, 1024);
    wt_transpose<<<(unsigned)(((size_t)512 * 512  + 255) / 256), 256, 0, stream>>>(w_vq,  wt_vq,  512, 512);
    wt_transpose<<<(unsigned)(((size_t)512 * 512  + 255) / 256), 256, 0, stream>>>(w_out, wt_out, 512, 512);

    // 3) big WMMA GEMMs on V rows (128x128 block tiles)
    gemm_bf16_wmma<<<dim3(MROWS_V / 128, 1024 / 128), 128, 0, stream>>>(
        xb, wt_vkv, vkv, nullptr, MROWS_V, 1024, DIM, 1024, /*a_mode=*/1, /*c_f32=*/0);
    gemm_bf16_wmma<<<dim3(MROWS_V / 128, 512 / 128), 128, 0, stream>>>(
        xb, wt_vq, vq, nullptr, MROWS_V, 512, DIM, 512, 1, 0);

    // 4) A-token projections (f32)
    aqkv_gemm<<<dim3(BT, 1536 / 256), 256, 0, stream>>>(x, w_aqkv, aqkv);

    // 5) attention 1 (A queries over 15680 V keys)
    att1_dots<<<dim3(BB * HH, (KV + 255) / 256), 256, 0, stream>>>(aqkv, vkv, dots);
    att1_softmax<<<BB * HH * TT, 256, 0, stream>>>(dots);
    att1_out<<<BB * HH, 320, 0, stream>>>(dots, vkv, proj);

    // 6) attention 2 (V queries over 5 A keys), masked, into proj_in
    att2_weights<<<(MROWS_V * HH) / 256, 256, 0, stream>>>(vq, aqkv, vaw);
    att2_out<<<MROWS_V, 512, 0, stream>>>(vaw, aqkv, mask, proj);

    // 7) output projection with bias -> d_out (f32)
    gemm_bf16_wmma<<<dim3((MROWS_O + 127) / 128, 512 / 128), 128, 0, stream>>>(
        proj, wt_out, out, b_out, MROWS_O, 512, DIM, 512, 0, /*c_f32=*/1);
}
